// MambaLow_62818191671579
// MI455X (gfx1250) — matrix operand order repaired and checked by reading, not compiled, not run on hardware
//
#include <hip/hip_runtime.h>
#include <math.h>

// ---- problem constants (match reference) ----
#define BATCH   2
#define SEQLEN  1024
#define BL      (BATCH*SEQLEN)     // 2048
#define DMODEL  512
#define DINNER  512
#define DHALF   256
#define DQ      128
#define DSTATE  128
#define DTRANK  32
#define XDBLW   (DTRANK + 2*DSTATE)   // 288
#define RANK    33
#define NITER   4

typedef _Float16 v16h __attribute__((ext_vector_type(16)));
typedef float    v8f  __attribute__((ext_vector_type(8)));

// ---------------------------------------------------------------------------
// Generic WMMA GEMM:  C[M,N] = opA(A) * opB(B) (+ bias[j])
//   TA=0: A is MxK row-major (A[i*lda+k]);  TA=1: A(i,k) = A[k*lda+i]
//   TB=0: B is NxK row-major "weight style" (B(k,j) = B[j*ldb+k])
//   TB=1: B is KxN row-major (B(k,j) = B[k*ldb+j])
// One wave computes a 32x64 block: 2 A-fragments x 4 B-fragments
// -> 8 v_wmma_f32_16x16x32_f16 per K-step (A reused 4x, B reused 2x).
// Partial N supertiles handled with wave-uniform 16-wide tile guards
// (EXEC stays all-ones for every issued WMMA, per ISA 7.12 restriction).
// Requires M%32==0, N%16==0, K%32==0 (true for all uses here).
// ---------------------------------------------------------------------------
template <int TA, int TB>
__global__ void gemm_wmma_k(const float* __restrict__ A, int lda,
                            const float* __restrict__ B, int ldb,
                            const float* __restrict__ bias,
                            float* __restrict__ C, int ldc,
                            int M, int N, int K)
{
    const int lane = threadIdx.x & 31;
    const int wave = threadIdx.x >> 5;
    const int nJS  = (N + 63) >> 6;                // 64-wide supertiles (may be partial)
    const int s    = blockIdx.x * 8 + wave;
    const int tot  = (M >> 5) * nJS;
    if (s >= tot) return;                          // wave-uniform
    const int i0 = (s / nJS) << 5;                 // 32-row block
    const int j0 = (s % nJS) << 6;                 // 64-col block
    const int m  = lane & 15;                      // A row; B and C/D column within tile
    const int h  = lane >> 4;                      // lane half

    v8f acc[2][4];
#pragma unroll
    for (int ii = 0; ii < 2; ++ii)
#pragma unroll
        for (int jt = 0; jt < 4; ++jt) acc[ii][jt] = (v8f){};

    for (int kb = 0; kb < K; kb += 32) {
        // Two A fragments: 16x32 f16 each (ISA 7.12.2 layout). Lane (m,h);
        // group g in {0,1}, t in [0,8): K = kb + g*16 + h*8 + t (ISA table order).
        v16h afr[2];
#pragma unroll
        for (int ii = 0; ii < 2; ++ii) {
            const int row = i0 + ii * 16 + m;
#pragma unroll
            for (int g = 0; g < 2; ++g)
#pragma unroll
                for (int t = 0; t < 8; ++t) {
                    const int k = kb + g * 16 + h * 8 + t;
                    const float v = TA ? A[(size_t)k * lda + row]
                                       : A[(size_t)row * lda + k];
                    afr[ii][g * 8 + t] = (_Float16)v;
                }
        }
        // Four B fragments (32x16 f16): lane column n = m, K = kb + (t/8)*16 + h*8 + t%8.
#pragma unroll
        for (int jt = 0; jt < 4; ++jt) {
            const int jb = j0 + jt * 16;
            if (jb < N) {                          // wave-uniform tile guard
                const int j = jb + m;
                v16h bf;
#pragma unroll
                for (int t = 0; t < 16; ++t) {
                    const int k = kb + (t >> 3) * 16 + h * 8 + (t & 7);
                    const float v = TB ? B[(size_t)k * ldb + j]
                                       : B[(size_t)j * ldb + k];
                    bf[t] = (_Float16)v;
                }
                acc[0][jt] = __builtin_amdgcn_wmma_f32_16x16x32_f16(
                    false, afr[0], false, bf, (short)0, acc[0][jt], false, false);
                acc[1][jt] = __builtin_amdgcn_wmma_f32_16x16x32_f16(
                    false, afr[1], false, bf, (short)0, acc[1][jt], false, false);
            }
        }
    }
    // C/D layout: VGPR r, lanes 0-15 -> M=r, lanes 16-31 -> M=8+r.
#pragma unroll
    for (int ii = 0; ii < 2; ++ii)
#pragma unroll
        for (int jt = 0; jt < 4; ++jt) {
            const int jb = j0 + jt * 16;
            if (jb < N) {
                const int j = jb + m;
                const float bv = bias ? bias[j] : 0.0f;
#pragma unroll
                for (int r = 0; r < 8; ++r)
                    C[(size_t)(i0 + ii * 16 + h * 8 + r) * ldc + j] = acc[ii][jt][r] + bv;
            }
        }
}

// ---------------------------------------------------------------------------
// Depthwise conv1d k=4, padding (1,2) ('same'), fused SiLU.
// src = xz (BL x 512), channel block at src_off; dst column block at dst_off.
// ---------------------------------------------------------------------------
__global__ void dwconv_silu_k(const float* __restrict__ src, int src_off,
                              const float* __restrict__ w,
                              float* __restrict__ dst, int dst_ld, int dst_off)
{
    const int t = blockIdx.x * blockDim.x + threadIdx.x;
    if (t >= BL * DHALF) return;
    const int c  = t & (DHALF - 1);
    const int bl = t >> 8;
    const int b  = bl / SEQLEN, l = bl % SEQLEN;
    float s = 0.0f;
#pragma unroll
    for (int q = 0; q < 4; ++q) {
        const int li = l + q - 1;
        if (li >= 0 && li < SEQLEN)
            s += w[c * 4 + q] * src[(size_t)(b * SEQLEN + li) * DINNER + src_off + c];
    }
    dst[(size_t)bl * dst_ld + dst_off + c] = s / (1.0f + __expf(-s));
}

// ---- min/max reduction over Amat (single workgroup) ----
__global__ void minmax_k(const float* __restrict__ A, int n, float* __restrict__ mm)
{
    __shared__ float smn[256], smx[256];
    float mn = 3.4e38f, mx = -3.4e38f;
    for (int i = threadIdx.x; i < n; i += 256) {
        const float v = A[i];
        mn = fminf(mn, v); mx = fmaxf(mx, v);
    }
    smn[threadIdx.x] = mn; smx[threadIdx.x] = mx; __syncthreads();
    for (int s = 128; s > 0; s >>= 1) {
        if ((int)threadIdx.x < s) {
            smn[threadIdx.x] = fminf(smn[threadIdx.x], smn[threadIdx.x + s]);
            smx[threadIdx.x] = fmaxf(smx[threadIdx.x], smx[threadIdx.x + s]);
        }
        __syncthreads();
    }
    if (threadIdx.x == 0) { mm[0] = smn[0]; mm[1] = smx[0]; }
}

__global__ void norm_noise_k(const float* __restrict__ A, const float* __restrict__ mm,
                             const float* __restrict__ noise, const float* __restrict__ eps,
                             float* __restrict__ out, int n)
{
    const int t = blockIdx.x * blockDim.x + threadIdx.x;
    if (t >= n) return;
    const float mn = mm[0], mx = mm[1];
    out[t] = (A[t] - mn) / (mx - mn) + noise[t] * eps[0];
}

// ---------------------------------------------------------------------------
// Low-rank (SVD-truncation) replacement: block power iteration on G = M^T M.
// 128 threads, one workgroup; Q (128 x RANK) kept in LDS; MGS orthonormalize.
// ---------------------------------------------------------------------------
__device__ void mgs128(float* Qs, float* red, int tid)
{
    for (int j = 0; j < RANK; ++j) {
        for (int i = 0; i < j; ++i) {
            red[tid] = Qs[tid * RANK + j] * Qs[tid * RANK + i];
            __syncthreads();
            for (int s = 64; s > 0; s >>= 1) { if (tid < s) red[tid] += red[tid + s]; __syncthreads(); }
            const float dot = red[0];
            __syncthreads();
            Qs[tid * RANK + j] -= dot * Qs[tid * RANK + i];
            __syncthreads();
        }
        red[tid] = Qs[tid * RANK + j] * Qs[tid * RANK + j];
        __syncthreads();
        for (int s = 64; s > 0; s >>= 1) { if (tid < s) red[tid] += red[tid + s]; __syncthreads(); }
        const float inv = rsqrtf(red[0] + 1e-20f);
        __syncthreads();
        Qs[tid * RANK + j] *= inv;
        __syncthreads();
    }
}

__global__ void subspace_k(const float* __restrict__ G, float* __restrict__ Q)
{
    __shared__ float Qs[128 * RANK];
    __shared__ float red[128];
    const int tid = threadIdx.x;
    for (int j = 0; j < RANK; ++j) Qs[tid * RANK + j] = G[tid * 128 + j];
    __syncthreads();
    for (int it = 0; it < NITER; ++it) {
        mgs128(Qs, red, tid);
        float tmp[RANK];
        for (int j = 0; j < RANK; ++j) tmp[j] = 0.0f;
        for (int k = 0; k < 128; ++k) {
            const float g = G[tid * 128 + k];
            for (int j = 0; j < RANK; ++j) tmp[j] += g * Qs[k * RANK + j];
        }
        __syncthreads();
        for (int j = 0; j < RANK; ++j) Qs[tid * RANK + j] = tmp[j];
        __syncthreads();
    }
    mgs128(Qs, red, tid);
    for (int j = 0; j < RANK; ++j) Q[tid * RANK + j] = Qs[tid * RANK + j];
}

// T (R x RANK) = M (R x 128, leading dim ldm) * Q (128 x RANK)
__global__ void mq_k(const float* __restrict__ Mx, int ldm, int R,
                     const float* __restrict__ Q, float* __restrict__ T)
{
    const int t = blockIdx.x * 256 + threadIdx.x;
    if (t >= R * RANK) return;
    const int j = t % RANK, r = t / RANK;
    float s = 0.0f;
    for (int k = 0; k < 128; ++k) s += Mx[(size_t)r * ldm + k] * Q[k * RANK + j];
    T[(size_t)r * RANK + j] = s;
}

// Out (R x 128) = T (R x RANK) * Q^T
__global__ void proj_k(const float* __restrict__ T, int R,
                       const float* __restrict__ Q, float* __restrict__ Out, int ldo)
{
    const int t = blockIdx.x * 256 + threadIdx.x;
    if (t >= R * 128) return;
    const int n = t & 127, r = t >> 7;
    float s = 0.0f;
    for (int j = 0; j < RANK; ++j) s += T[(size_t)r * RANK + j] * Q[n * RANK + j];
    Out[(size_t)r * ldo + n] = s;
}

// ---------------------------------------------------------------------------
// Selective scan: one wave per (b, d); 4 states per lane (n = lane + 32q);
// wave32 shuffle reduction for the C contraction each step.
// delta = softplus(dt + b_dt)  (dt already carries one b_dt; reference adds it twice)
// Next-step B/C rows are prefetched (global_prefetch_b8) to hide L2 latency
// behind the serial exp/shuffle chain.
// ---------------------------------------------------------------------------
__global__ void scan_k(const float* __restrict__ dt, const float* __restrict__ u,
                       const float* __restrict__ A,  const float* __restrict__ Bm,
                       const float* __restrict__ Cm, const float* __restrict__ Dv,
                       const float* __restrict__ bdt, float* __restrict__ y)
{
    const int lane = threadIdx.x & 31;
    const int wid  = (blockIdx.x * blockDim.x + threadIdx.x) >> 5;
    const int b = wid >> 8, d = wid & 255;
    float a[4], st[4];
#pragma unroll
    for (int q = 0; q < 4; ++q) { a[q] = A[(size_t)d * DSTATE + lane + 32 * q]; st[q] = 0.0f; }
    const float Dd = Dv[d], bd = bdt[d];
    for (int l = 0; l < SEQLEN; ++l) {
        const int bl = b * SEQLEN + l;
        if (l + 1 < SEQLEN) {
            __builtin_prefetch(&Bm[(size_t)(bl + 1) * DSTATE + lane], 0, 0);
            __builtin_prefetch(&Cm[(size_t)(bl + 1) * DSTATE + lane], 0, 0);
        }
        const float dv = dt[(size_t)bl * DHALF + d] + bd;
        const float delta = dv > 20.0f ? dv : log1pf(__expf(dv));
        const float ul = u[(size_t)bl * DHALF + d];
        const float du = delta * ul;
        float acc = 0.0f;
#pragma unroll
        for (int q = 0; q < 4; ++q) {
            const int n = lane + 32 * q;
            const float bn = Bm[(size_t)bl * DSTATE + n];
            const float cn = Cm[(size_t)bl * DSTATE + n];
            st[q] = __expf(delta * a[q]) * st[q] + du * bn;
            acc += st[q] * cn;
        }
        for (int off = 16; off; off >>= 1) acc += __shfl_xor(acc, off, 32);
        if (lane == 0) y[(size_t)bl * DINNER + d] = acc + ul * Dd;
    }
}

// ---------------------------------------------------------------------------
static inline int gemm_blocks(int M, int N)
{
    return (((M / 32) * ((N + 63) / 64)) + 7) / 8;
}

extern "C" void kernel_launch(void* const* d_in, const int* in_sizes, int n_in,
                              void* d_out, int out_size, void* d_ws, size_t ws_size,
                              hipStream_t stream)
{
    const float* hidden  = (const float*)d_in[0];   // (2048, 512)
    const float* W_in    = (const float*)d_in[1];   // (512, 512)
    const float* W_getA  = (const float*)d_in[2];   // (128, 256)
    const float* b_getA  = (const float*)d_in[3];   // (128)
    const float* cw_x    = (const float*)d_in[4];   // (256,1,4)
    const float* cw_z    = (const float*)d_in[5];   // (256,1,4)
    const float* W_xproj = (const float*)d_in[6];   // (288, 256)
    const float* W_dt    = (const float*)d_in[7];   // (256, 32)
    const float* b_dt    = (const float*)d_in[8];   // (256)
    const float* Dvec    = (const float*)d_in[9];   // (256)
    const float* eps     = (const float*)d_in[10];  // (1)
    const float* W_out   = (const float*)d_in[11];  // (512, 512)
    const float* noise   = (const float*)d_in[12];  // (256, 128)
    float* out = (float*)d_out;

    // workspace layout (floats)
    float* W = (float*)d_ws;
    size_t o = 0;
    float* xz    = W + o; o += (size_t)BL * DINNER;      // 2048x512
    float* getA2 = W + o; o += (size_t)BL * DQ;          // 2048x128
    float* Amat  = W + o; o += (size_t)DHALF * DQ;       // 256x128
    float* adjA  = W + o; o += (size_t)DHALF * DQ;       // 256x128
    float* xc    = W + o; o += (size_t)BL * DHALF;       // 2048x256 (conv'd x)
    float* ycat  = W + o; o += (size_t)BL * DINNER;      // 2048x512 (y | z)
    float* xdbl  = W + o; o += (size_t)BL * XDBLW;       // 2048x288
    float* dtb   = W + o; o += (size_t)BL * DHALF;       // 2048x256
    float* G     = W + o; o += (size_t)DQ * DQ;          // 128x128
    float* Q     = W + o; o += (size_t)DQ * RANK;        // 128x33
    float* T     = W + o; o += (size_t)BL * RANK;        // 2048x33 (max)
    float* Alr   = W + o; o += (size_t)DHALF * DQ;       // 256x128
    float* Blr   = W + o; o += (size_t)BL * DQ;          // 2048x128
    float* Clr   = W + o; o += (size_t)BL * DQ;          // 2048x128
    float* mm    = W + o; o += 2;

    // 1. xz = hidden @ W_in^T
    gemm_wmma_k<0,0><<<gemm_blocks(BL, DINNER), 256, 0, stream>>>(
        hidden, DMODEL, W_in, DMODEL, nullptr, xz, DINNER, BL, DINNER, DMODEL);

    // 2. getA2 = x_pre @ W_getA^T + b_getA   (x_pre = xz[:, :256])
    gemm_wmma_k<0,0><<<gemm_blocks(BL, DQ), 256, 0, stream>>>(
        xz, DINNER, W_getA, DHALF, b_getA, getA2, DQ, BL, DQ, DHALF);

    // 3. Amat = x_pre^T @ getA2   (256 x 128, K = 2048)
    gemm_wmma_k<1,1><<<gemm_blocks(DHALF, DQ), 256, 0, stream>>>(
        xz, DINNER, getA2, DQ, nullptr, Amat, DQ, DHALF, DQ, BL);

    // 4/5. min-max normalize + noise*eps
    minmax_k<<<1, 256, 0, stream>>>(Amat, DHALF * DQ, mm);
    norm_noise_k<<<(DHALF * DQ + 255) / 256, 256, 0, stream>>>(
        Amat, mm, noise, eps, adjA, DHALF * DQ);

    // 6. depthwise conv + SiLU (x -> xc; z -> ycat[:, 256:512])
    dwconv_silu_k<<<(BL * DHALF + 255) / 256, 256, 0, stream>>>(xz, 0,     cw_x, xc,   DHALF,  0);
    dwconv_silu_k<<<(BL * DHALF + 255) / 256, 256, 0, stream>>>(xz, DHALF, cw_z, ycat, DINNER, DHALF);

    // 7. x_dbl = xc @ W_xproj^T  (2048 x 288, last 64-wide supertile partial)
    gemm_wmma_k<0,0><<<gemm_blocks(BL, XDBLW), 256, 0, stream>>>(
        xc, DHALF, W_xproj, DHALF, nullptr, xdbl, XDBLW, BL, XDBLW, DHALF);

    // 8. dt = dt_r @ W_dt^T + b_dt   (dt_r = x_dbl[:, :32])
    gemm_wmma_k<0,0><<<gemm_blocks(BL, DHALF), 256, 0, stream>>>(
        xdbl, XDBLW, W_dt, DTRANK, b_dt, dtb, DHALF, BL, DHALF, DTRANK);

    // 9. low-rank A: G = adjA^T adjA -> subspace Q -> Alr = (adjA Q) Q^T
    gemm_wmma_k<1,1><<<gemm_blocks(DQ, DQ), 256, 0, stream>>>(
        adjA, DQ, adjA, DQ, nullptr, G, DQ, DQ, DQ, DHALF);
    subspace_k<<<1, 128, 0, stream>>>(G, Q);
    mq_k<<<(DHALF * RANK + 255) / 256, 256, 0, stream>>>(adjA, DQ, DHALF, Q, T);
    proj_k<<<(DHALF * DQ + 255) / 256, 256, 0, stream>>>(T, DHALF, Q, Alr, DQ);

    // 10. low-rank B: Bp = x_dbl[:, 32:160]
    gemm_wmma_k<1,1><<<gemm_blocks(DQ, DQ), 256, 0, stream>>>(
        xdbl + DTRANK, XDBLW, xdbl + DTRANK, XDBLW, nullptr, G, DQ, DQ, DQ, BL);
    subspace_k<<<1, 128, 0, stream>>>(G, Q);
    mq_k<<<(BL * RANK + 255) / 256, 256, 0, stream>>>(xdbl + DTRANK, XDBLW, BL, Q, T);
    proj_k<<<(BL * DQ + 255) / 256, 256, 0, stream>>>(T, BL, Q, Blr, DQ);

    // 11. low-rank C: Cp = x_dbl[:, 160:288]
    gemm_wmma_k<1,1><<<gemm_blocks(DQ, DQ), 256, 0, stream>>>(
        xdbl + DTRANK + DSTATE, XDBLW, xdbl + DTRANK + DSTATE, XDBLW, nullptr, G, DQ, DQ, DQ, BL);
    subspace_k<<<1, 128, 0, stream>>>(G, Q);
    mq_k<<<(BL * RANK + 255) / 256, 256, 0, stream>>>(xdbl + DTRANK + DSTATE, XDBLW, BL, Q, T);
    proj_k<<<(BL * DQ + 255) / 256, 256, 0, stream>>>(T, BL, Q, Clr, DQ);

    // 12. selective scan -> ycat[:, 0:256]
    scan_k<<<64, 256, 0, stream>>>(dtb, xc, Alr, Blr, Clr, Dvec, b_dt, ycat);

    // 13. out = ycat @ W_out^T
    gemm_wmma_k<0,0><<<gemm_blocks(BL, DMODEL), 256, 0, stream>>>(
        ycat, DINNER, W_out, DINNER, nullptr, out, DMODEL, BL, DMODEL, DINNER);
}
